// CGCNN_60181081752148
// MI455X (gfx1250) — matrix operand order repaired and checked by reading, not compiled, not run on hardware
//
#include <hip/hip_runtime.h>

#define NN 50000
#define EE 800000
#define GG 256
#define NF 16
#define EFD 8
#define HH 128
#define LL 4
#define EPS 1e-5f

#define KT 9        // k-tiles of 32 (264 padded to 288)
#define NT 8        // n-tiles of 16 (H=128)
#define MT 4        // m-tiles per block
#define MBLK (MT*16)       // 64 edges per block
#define WMAT (KT*NT*512)   // bf16 elems per packed matrix = 36864

typedef __attribute__((ext_vector_type(16))) __bf16 v16bf;
typedef __attribute__((ext_vector_type(8)))  float  v8f;

static __device__ __forceinline__ v8f wmma_bf16(v16bf a, v16bf b, v8f c) {
  return __builtin_amdgcn_wmma_f32_16x16x32_bf16(
      /*neg_a=*/false, a, /*neg_b=*/false, b,
      /*c_mod=*/(short)0, c, /*reuse_a=*/false, /*reuse_b=*/false);
}

// gfx1250 async direct-to-LDS copy (ASYNCcnt-tracked; no VGPR round trip)
static __device__ __forceinline__ void async_gather_b128(unsigned lds_off,
                                                         const void* gaddr) {
  asm volatile("global_load_async_to_lds_b128 %0, %1, off"
               :: "v"(lds_off), "v"(gaddr)
               : "memory");
}
static __device__ __forceinline__ void wait_asynccnt0() {
  asm volatile("s_wait_asynccnt 0x0" ::: "memory");
}

static __device__ __forceinline__ unsigned short f2bf(float f) {
  union { float f; unsigned int u; } v; v.f = f;
  unsigned int u = v.u;
  u += 0x7fffu + ((u >> 16) & 1u);   // round-to-nearest-even
  return (unsigned short)(u >> 16);
}

// ---------------- weight packing into WMMA B-operand lane layout ----------------
// wpack layout: [mat(8 = L*2)][kt(9)][nt(8)][lane(32)][16 bf16]
// B 32x16 bf16 order chosen here (A matches): lane l<16: col n=l, elems 0..15 = K kb..kb+15
//                      lane l+16: col n=l, elems 0..15 = K kb+16..kb+31
__global__ __launch_bounds__(256) void pack_w_kernel(
    const float* __restrict__ Wf, const float* __restrict__ Ws,
    unsigned short* __restrict__ wpack) {
  int i = blockIdx.x * 256 + threadIdx.x;
  if (i >= 8 * WMAT) return;
  int p    = i & 15;
  int lane = (i >> 4) & 31;
  int nt   = (i >> 9) & 7;
  int q    = i >> 12;          // mat*9 + kt
  int kt   = q % 9;
  int mat  = q / 9;
  int l    = mat >> 1;
  bool isS = mat & 1;
  int k = kt * 32 + (lane >> 4) * 16 + p;
  int n = nt * 16 + (lane & 15);
  float v = 0.0f;
  if (k < 2 * HH + EFD) v = (isS ? Ws : Wf)[(l * (2 * HH + EFD) + k) * HH + n];
  wpack[i] = f2bf(v);
}

__global__ __launch_bounds__(256) void ecvt_kernel(const float* __restrict__ ea,
                                                   unsigned short* __restrict__ ebf) {
  int i = blockIdx.x * 256 + threadIdx.x;
  if (i < EE * EFD) ebf[i] = f2bf(ea[i]);
}

// ---------------- node embedding: h = relu(x @ W_emb + b) ----------------
__global__ __launch_bounds__(256) void embed_kernel(
    const float* __restrict__ x, const float* __restrict__ Wemb,
    const float* __restrict__ bemb, float* __restrict__ h,
    unsigned short* __restrict__ hbf) {
  int i = blockIdx.x * 256 + threadIdx.x;
  if (i >= NN * HH) return;
  int n = i >> 7, c = i & 127;
  float acc = bemb[c];
#pragma unroll
  for (int k = 0; k < NF; ++k) acc += x[n * NF + k] * Wemb[k * HH + c];
  acc = fmaxf(acc, 0.0f);
  h[i] = acc;
  hbf[i] = f2bf(acc);
}

// ---------------- fused gather + dual-GEMM (WMMA bf16) + gate + scatter ----------------
// Block: 256 threads = 8 waves. Block owns 64 edges (4 m-tiles); wave w owns n-tile w.
// All 9 k-tiles of the A operand are staged into LDS in the exact WMMA A lane
// layout via async-to-LDS gathers, then a single barrier precedes the WMMA loop.
__global__ __launch_bounds__(256) void edge_gemm_kernel(
    const unsigned short* __restrict__ hbf,
    const unsigned short* __restrict__ ebf,
    const unsigned short* __restrict__ wpF,
    const unsigned short* __restrict__ wpS,
    const float* __restrict__ biasF, const float* __restrict__ biasS,
    const int* __restrict__ esrc, const int* __restrict__ edst,
    float* __restrict__ agg) {
  // A staging: [kt][mtile][lane slot][2 x 16B chunks] = 36 KB
  __shared__ __align__(32) uint4 As4[KT][MT][32][2];

  const int t = threadIdx.x;
  const int eBase = blockIdx.x * MBLK;

  // every thread stages one (mtile, lane-slot, chunk) across all k-tiles
  const int lmt   = t >> 6;         // 0..3
  const int lrem  = t & 63;
  const int lslot = lrem >> 1;      // lane slot 0..31
  const int lj    = lrem & 1;       // chunk 0/1
  const int lhalf = lslot >> 4;
  const int le    = eBase + lmt * 16 + (lslot & 15);
  const int lSrc  = esrc[le];
  const int lDst  = edst[le];
  // A 16x32 bf16 K order chosen here (matches B): slot s: row s&15, chunk K = kb + (s>>4)*16 + j*8
  const int cOff  = lhalf * 16 + lj * 8;

  const unsigned short* pD = hbf + lDst * HH + cOff;   // x_i = h[dst], K 0..127
  const unsigned short* pS = hbf + lSrc * HH + cOff;   // x_j = h[src], K 128..255
#pragma unroll
  for (int kt = 0; kt < 4; ++kt)
    async_gather_b128((unsigned)(uintptr_t)&As4[kt][lmt][lslot][lj], pD + kt * 32);
#pragma unroll
  for (int kt = 4; kt < 8; ++kt)
    async_gather_b128((unsigned)(uintptr_t)&As4[kt][lmt][lslot][lj], pS + (kt - 4) * 32);
  if (cOff == 0) {   // K 256..263 = edge_attr; remaining pad chunks are zero
    async_gather_b128((unsigned)(uintptr_t)&As4[8][lmt][lslot][0], ebf + le * EFD);
  } else {
    uint4 z = {0u, 0u, 0u, 0u};
    As4[8][lmt][lslot][lj] = z;
  }
  wait_asynccnt0();
  __syncthreads();

  const int wid  = t >> 5;          // wave id = n-tile
  const int lane = t & 31;

  v8f zero = {0.f, 0.f, 0.f, 0.f, 0.f, 0.f, 0.f, 0.f};
  v8f accF[MT], accS[MT];
#pragma unroll
  for (int mt = 0; mt < MT; ++mt) { accF[mt] = zero; accS[mt] = zero; }

#pragma unroll
  for (int kt = 0; kt < KT; ++kt) {
    const v16bf bF = *(const v16bf*)(wpF + (kt * NT + wid) * 512 + lane * 16);
    const v16bf bS = *(const v16bf*)(wpS + (kt * NT + wid) * 512 + lane * 16);
#pragma unroll
    for (int mt = 0; mt < MT; ++mt) {
      const v16bf a = *(const v16bf*)(&As4[kt][mt][lane][0]);
      accF[mt] = wmma_bf16(a, bF, accF[mt]);
      accS[mt] = wmma_bf16(a, bS, accS[mt]);
    }
  }

  // C/D layout: elem r -> row (r + 8*half), col = lane&15 (within this wave's n-tile)
  const int half = lane >> 4;
  const int n = wid * 16 + (lane & 15);
  const float bFv = biasF[n], bSv = biasS[n];
#pragma unroll
  for (int mt = 0; mt < MT; ++mt) {
#pragma unroll
    for (int r = 0; r < 8; ++r) {
      int e2 = eBase + mt * 16 + r + 8 * half;
      float fv = accF[mt][r] + bFv;
      float sv = accS[mt][r] + bSv;
      float sig = 1.0f / (1.0f + __expf(-fv));
      float sp  = (sv > 15.0f) ? sv : log1pf(__expf(sv));
      float gate = sig * sp;
      int dd = edst[e2];
      unsafeAtomicAdd(&agg[dd * HH + n], gate);
    }
  }
}

// ---------------- batch-norm statistics (sum, sumsq per channel) ----------------
__global__ __launch_bounds__(256) void bn_stats_kernel(
    const float* __restrict__ agg, const float* __restrict__ h,
    float* __restrict__ stats) {
  int c = threadIdx.x & 127;
  int n0 = blockIdx.x * 2 + (threadIdx.x >> 7);
  float s = 0.0f, s2 = 0.0f;
  for (int n = n0; n < NN; n += gridDim.x * 2) {
    float v = agg[n * HH + c] + h[n * HH + c];
    s += v;
    s2 += v * v;
  }
  unsafeAtomicAdd(&stats[c], s);
  unsafeAtomicAdd(&stats[HH + c], s2);
}

// ---------------- BN + relu + residual; writes fp32 h and bf16 mirror ----------------
__global__ __launch_bounds__(256) void bn_update_kernel(
    const float* __restrict__ agg, float* __restrict__ h,
    unsigned short* __restrict__ hbf, const float* __restrict__ stats,
    const float* __restrict__ gamma, const float* __restrict__ beta) {
  int i = blockIdx.x * 256 + threadIdx.x;
  if (i >= NN * HH) return;
  int c = i & 127;
  const float invN = 1.0f / (float)NN;
  float mu = stats[c] * invN;
  float var = stats[HH + c] * invN - mu * mu;
  float v = agg[i] + h[i];
  float xn = gamma[c] * (v - mu) * rsqrtf(var + EPS) + beta[c];
  xn = fmaxf(xn, 0.0f);
  float hn = h[i] + xn;
  h[i] = hn;
  hbf[i] = f2bf(hn);
}

// ---------------- global mean pool (atomic accumulate) ----------------
__global__ __launch_bounds__(256) void pool_kernel(
    const float* __restrict__ h, const int* __restrict__ batch,
    float* __restrict__ pooled, float* __restrict__ cnt) {
  int i = blockIdx.x * 256 + threadIdx.x;
  if (i >= NN * HH) return;
  int n = i >> 7, c = i & 127;
  int b = batch[n];
  unsafeAtomicAdd(&pooled[b * HH + c], h[i]);
  if (c == 0) unsafeAtomicAdd(&cnt[b], 1.0f);
}

// ---------------- tiny MLP head: one thread per graph ----------------
__global__ __launch_bounds__(256) void head_kernel(
    const float* __restrict__ pooled, const float* __restrict__ cnt,
    const float* __restrict__ W1, const float* __restrict__ b1,
    const float* __restrict__ W2, const float* __restrict__ b2,
    const float* __restrict__ Wout, const float* __restrict__ bout,
    float* __restrict__ out) {
  int g = threadIdx.x;
  float inv = 1.0f / fmaxf(cnt[g], 1.0f);
  float g1[64];
#pragma unroll 4
  for (int j = 0; j < 64; ++j) {
    float acc = b1[j];
    for (int k = 0; k < HH; ++k) acc += pooled[g * HH + k] * inv * W1[k * 64 + j];
    g1[j] = fmaxf(acc, 0.0f);
  }
  float g2[32];
#pragma unroll 4
  for (int j = 0; j < 32; ++j) {
    float acc = b2[j];
    for (int k = 0; k < 64; ++k) acc += g1[k] * W2[k * 32 + j];
    g2[j] = fmaxf(acc, 0.0f);
  }
  float acc = bout[0];
#pragma unroll
  for (int k = 0; k < 32; ++k) acc += g2[k] * Wout[k];
  out[g] = acc;
}

// =====================================================================
extern "C" void kernel_launch(void* const* d_in, const int* in_sizes, int n_in,
                              void* d_out, int out_size, void* d_ws, size_t ws_size,
                              hipStream_t stream) {
  const float* x     = (const float*)d_in[0];
  const int*   eidx  = (const int*)d_in[1];
  const int*   esrc  = eidx;
  const int*   edst  = eidx + EE;
  const float* eattr = (const float*)d_in[2];
  const int*   batch = (const int*)d_in[3];
  const float* Wemb  = (const float*)d_in[4];
  const float* bemb  = (const float*)d_in[5];
  const float* Wf    = (const float*)d_in[6];
  const float* bfv   = (const float*)d_in[7];
  const float* Ws    = (const float*)d_in[8];
  const float* bsv   = (const float*)d_in[9];
  const float* gamma = (const float*)d_in[10];
  const float* beta  = (const float*)d_in[11];
  const float* W1    = (const float*)d_in[12];
  const float* b1    = (const float*)d_in[13];
  const float* W2    = (const float*)d_in[14];
  const float* b2    = (const float*)d_in[15];
  const float* Wout  = (const float*)d_in[16];
  const float* bout  = (const float*)d_in[17];
  float* out = (float*)d_out;

  // workspace carving (256B aligned)
  char* ws = (char*)d_ws;
  size_t off = 0;
  auto carve = [&](size_t bytes) -> char* {
    char* p = ws + off;
    off = (off + bytes + 255) & ~(size_t)255;
    return p;
  };
  float*          h      = (float*)carve((size_t)NN * HH * 4);
  unsigned short* hbf    = (unsigned short*)carve((size_t)NN * HH * 2);
  float*          agg    = (float*)carve((size_t)NN * HH * 4);
  unsigned short* ebf    = (unsigned short*)carve((size_t)EE * EFD * 2);
  unsigned short* wpack  = (unsigned short*)carve((size_t)8 * WMAT * 2);
  float*          stats  = (float*)carve(2 * HH * 4);
  float*          pooled = (float*)carve((size_t)GG * HH * 4);
  float*          cnt    = (float*)carve((size_t)GG * 4);

  // prep: pack weights to WMMA B layout, bf16 edge attrs, embedding
  pack_w_kernel<<<(8 * WMAT + 255) / 256, 256, 0, stream>>>(Wf, Ws, wpack);
  ecvt_kernel<<<(EE * EFD + 255) / 256, 256, 0, stream>>>(eattr, ebf);
  embed_kernel<<<(NN * HH + 255) / 256, 256, 0, stream>>>(x, Wemb, bemb, h, hbf);

  for (int l = 0; l < LL; ++l) {
    hipMemsetAsync(agg, 0, (size_t)NN * HH * 4, stream);
    edge_gemm_kernel<<<EE / MBLK, 256, 0, stream>>>(
        hbf, ebf, wpack + (size_t)(2 * l) * WMAT, wpack + (size_t)(2 * l + 1) * WMAT,
        bfv + l * HH, bsv + l * HH, esrc, edst, agg);
    hipMemsetAsync(stats, 0, 2 * HH * 4, stream);
    bn_stats_kernel<<<256, 256, 0, stream>>>(agg, h, stats);
    bn_update_kernel<<<(NN * HH + 255) / 256, 256, 0, stream>>>(
        agg, h, hbf, stats, gamma + l * HH, beta + l * HH);
  }

  hipMemsetAsync(pooled, 0, (size_t)GG * HH * 4, stream);
  hipMemsetAsync(cnt, 0, (size_t)GG * 4, stream);
  pool_kernel<<<(NN * HH + 255) / 256, 256, 0, stream>>>(h, batch, pooled, cnt);
  head_kernel<<<1, 256, 0, stream>>>(pooled, cnt, W1, b1, W2, b2, Wout, bout, out);
}
